// RegionProposalNetwork_5669356833122
// MI455X (gfx1250) — matrix operand order repaired and checked by reading, not compiled, not run on hardware
//
#include <hip/hip_runtime.h>
#include <hip/hip_bf16.h>
#include <stdint.h>

// ---------------------------------------------------------------------------
// RPN for MI455X (gfx1250, wave32). Dominant cost: 3x3 conv 512->512 over
// 8x64x64 -> ~155 GFLOP at ~1000 FLOP/byte => matrix-op bound => bf16 WMMA,
// with the A tile streamed into LDS by the Tensor Data Mover (double-buffered,
// one tap ahead, tracked by TENSORcnt).
// ---------------------------------------------------------------------------

typedef __attribute__((ext_vector_type(16))) __bf16 v16bf;
typedef __attribute__((ext_vector_type(8)))  __bf16 v8bf;
typedef __attribute__((ext_vector_type(8)))  float  v8f;
typedef unsigned int u32x4 __attribute__((ext_vector_type(4)));
typedef int          i32x4 __attribute__((ext_vector_type(4)));
typedef int          i32x8 __attribute__((ext_vector_type(8)));

#define NB    8
#define CIN   512
#define CMID  512
#define HH    64
#define WW    64
#define HP    66
#define WP    66
#define NBOX  36864      // 64*64*9
#define NPOST 600

// Toolchain-detect TDM builtin arity: the gfx1250 TDM header ships only with
// the 6-arg (clang-23/therock) toolchain; ROCm 7.2 (clang-22) is 5-arg.
#if defined(__has_builtin)
# if __has_builtin(__builtin_amdgcn_tensor_load_to_lds)
#  define HAS_TDM 1
# else
#  define HAS_TDM 0
# endif
#else
# define HAS_TDM 0
#endif
#if defined(__has_include)
# if __has_include(<hip/amd_detail/amd_gfx1250_TDM.h>)
#  define TDM_SIX_ARGS 1
# else
#  define TDM_SIX_ARGS 0
# endif
#else
# define TDM_SIX_ARGS 0
#endif

__device__ __forceinline__ __bf16 to_bf16(float v) { return (__bf16)v; }

// ---------------------------------------------------------------------------
// Prep: x (f32 NCHW) -> xb (bf16 NHWC, 1-px zero halo: [N][66][66][512])
// ---------------------------------------------------------------------------
__global__ void prep_x_kernel(const float* __restrict__ x,
                              __bf16* __restrict__ xb, int total) {
  int i = blockIdx.x * blockDim.x + threadIdx.x;
  if (i >= total) return;
  int c = i & 511;
  int t = i >> 9;
  int xx = t % WP; t /= WP;
  int yy = t % HP; t /= HP;
  int n = t;
  float v = 0.f;
  if (xx >= 1 && xx <= WW && yy >= 1 && yy <= HH) {
    v = x[(((size_t)n * CIN + c) * HH + (yy - 1)) * WW + (xx - 1)];
  }
  xb[i] = to_bf16(v);
}

// ---------------------------------------------------------------------------
// Prep: w1 (f32 [O][I][3][3]) -> wt (bf16 [tap][out][in]); per-lane B frags
// become contiguous, 32B-aligned loads.
// ---------------------------------------------------------------------------
__global__ void prep_w_kernel(const float* __restrict__ w1,
                              __bf16* __restrict__ wt, int total) {
  int i = blockIdx.x * blockDim.x + threadIdx.x;
  if (i >= total) return;
  int in  = i & 511;
  int t   = i >> 9;
  int out = t & 511;
  int tap = t >> 9;
  int ky = tap / 3, kx = tap % 3;
  wt[i] = to_bf16(w1[(((size_t)out * CIN + in) * 3 + ky) * 3 + kx]);
}

// ---------------------------------------------------------------------------
// 3x3 conv as implicit GEMM with V_WMMA_F32_16X16X32_BF16.
// Block = 128 threads (4 waves). Tile: 16 positions x 256 out channels
// (64 per wave -> 4 v8f accumulators, A fragment reused 4x).
// The tap's A tile (16 pos x 512 ch, NHWC-contiguous 16KB) is DMA'd into a
// double-buffered LDS tile by the TDM, issued one tap ahead.
// ---------------------------------------------------------------------------
__global__ __launch_bounds__(128)
void conv3x3_wmma_kernel(const __bf16* __restrict__ xb,
                         const __bf16* __restrict__ wt,
                         const float*  __restrict__ b1,
                         __bf16* __restrict__ f) {
  int bid = blockIdx.x;
  int cg = bid & 1;  bid >>= 1;   // out-channel group (256 ch)
  int xt = bid & 3;  bid >>= 2;   // col tile (16 wide)
  int y  = bid & 63; bid >>= 6;   // row
  int n  = bid;                   // image
  int x0 = xt * 16;

  int tid  = threadIdx.x;
  int wave = tid >> 5;
  int lane = tid & 31;
  int m16  = lane & 15;
  int half = (lane >> 4) & 1;
  int half8 = half ? 8 : 0;

  int chbase = cg * 256 + wave * 64;     // this wave's 64 out channels
  const int ofsA = half ? 16 : 0;

  __shared__ __attribute__((aligned(64))) __bf16 As[2][16 * CIN];  // 2 x 16KB

  v8f acc[4] = {};

#if HAS_TDM
  // Wave-uniform TDM issue: 1-D contiguous tile of 8192 bf16 (16KB) -> LDS.
  auto tdm_issue = [&](int tap, int buf) {
    int ky = tap / 3, kx = tap - ky * 3;
    const __bf16* src =
        xb + (((size_t)n * HP + (y + ky)) * WP + (x0 + kx)) * (size_t)CIN;
    unsigned long long ga = (unsigned long long)(uintptr_t)src;
    unsigned int lds = (unsigned int)(uintptr_t)&As[buf][0];
    u32x4 g0;
    g0[0] = 1u;                                   // count=1, user mode
    g0[1] = lds;                                  // lds_addr
    g0[2] = (unsigned int)(ga & 0xFFFFFFFFu);     // global_addr[31:0]
    g0[3] = (unsigned int)((ga >> 32) & 0x01FFFFFFu) | (2u << 30); // type=2
    i32x8 g1;
    g1[0] = (1 << 16);                            // data_size=1 (2 bytes)
    g1[1] = (8192 & 0xFFFF) << 16;                // tensor_dim0[15:0]
    g1[2] = (1 << 16);                            // dim0 hi=0, tensor_dim1=1
    g1[3] = (8192 & 0xFFFF) << 16;                // dim1 hi=0, tile_dim0=8192
    g1[4] = 1;                                    // tile_dim1=1, tile_dim2=0
    g1[5] = 8192;                                 // tensor_dim0_stride lo
    g1[6] = 0;
    g1[7] = 0;
    i32x4 gz = {0, 0, 0, 0};
#if TDM_SIX_ARGS
    i32x8 z8 = {0, 0, 0, 0, 0, 0, 0, 0};
    __builtin_amdgcn_tensor_load_to_lds(g0, g1, gz, gz, z8, 0);
#else
    __builtin_amdgcn_tensor_load_to_lds(g0, g1, gz, gz, 0);
#endif
  };
  if (wave == 0) tdm_issue(0, 0);
#endif

  for (int tap = 0; tap < 9; ++tap) {
#if HAS_TDM
    int buf = tap & 1;
    if (wave == 0) {
      if (tap + 1 < 9) {
        tdm_issue(tap + 1, buf ^ 1);              // prefetch next tap
        __builtin_amdgcn_s_wait_tensorcnt(1);     // wait: this tap's DMA done
      } else {
        __builtin_amdgcn_s_wait_tensorcnt(0);     // last tap: drain
      }
    }
    __syncthreads();                       // LDS tile for this tap visible
#else
    int buf = 0;
    {
      int ky = tap / 3, kx = tap - ky * 3;
      const __bf16* src =
          xb + (((size_t)n * HP + (y + ky)) * WP + (x0 + kx)) * (size_t)CIN;
      int e = tid * 64;                    // 8192 el / 128 thr = 64 each
#pragma unroll
      for (int k = 0; k < 64; k += 8)
        *(v8bf*)(&As[0][e + k]) = *(const v8bf*)(src + e + k);
    }
    __syncthreads();
#endif

    const __bf16* arow = &As[buf][m16 * CIN];
    const __bf16* w0 =
        wt + ((size_t)tap * CMID + chbase + m16) * (size_t)CIN;

    for (int kc = 0; kc < CIN; kc += 32) {
      // A fragment (lanes 0-15: K{0..15}; lanes 16-31: K{16..31})
      v16bf afrag;
      {
        v8bf lo = *(const v8bf*)(arow + kc + ofsA);
        v8bf hi = *(const v8bf*)(arow + kc + 8 + ofsA);
#pragma unroll
        for (int r = 0; r < 8; ++r) { afrag[r] = lo[r]; afrag[r + 8] = hi[r]; }
      }
      __builtin_prefetch(w0 + kc + 64, 0, 3);        // weight stream, WGP scope
#pragma unroll
      for (int j = 0; j < 4; ++j) {
        v16bf bfrag =
            *(const v16bf*)(w0 + (size_t)j * 16 * CIN + kc + (half ? 16 : 0));
        acc[j] = __builtin_amdgcn_wmma_f32_16x16x32_bf16(
            false, afrag, false, bfrag, (short)0, acc[j], false, false);
      }
    }
    __syncthreads();                       // done reading buf before reuse
  }

  // Epilogue: +bias, ReLU, store bf16 NHWC. C layout: VGPR r -> M=r+8*half.
#pragma unroll
  for (int j = 0; j < 4; ++j) {
    int col = chbase + j * 16 + m16;
    float bias = b1[col];
#pragma unroll
    for (int r = 0; r < 8; ++r) {
      int mrow = r + half8;
      float v = acc[j][r] + bias;
      v = v > 0.f ? v : 0.f;
      size_t oidx =
          (((size_t)n * HH + y) * WW + (x0 + mrow)) * (size_t)CMID + col;
      f[oidx] = to_bf16(v);
    }
  }
}

// ---------------------------------------------------------------------------
// 1x1 heads: 54 output channels (36 loc + 18 score); ~1.9 GFLOP -> VALU ok.
// NHWC position order == reference transpose(0,2,3,1).reshape order.
// ---------------------------------------------------------------------------
__global__ void head1x1_kernel(const __bf16* __restrict__ f,
                               const float* __restrict__ w_loc,
                               const float* __restrict__ b_loc,
                               const float* __restrict__ w_score,
                               const float* __restrict__ b_score,
                               float* __restrict__ out_loc,
                               float* __restrict__ out_score) {
  int idx = blockIdx.x * blockDim.x + threadIdx.x;
  int o  = idx & 63;
  int pg = idx >> 6;               // n*4096 + p
  if (pg >= NB * HH * WW) return;
  if (o >= 54) return;
  int n = pg >> 12;
  int p = pg & 4095;

  const __bf16* fb = f + (size_t)pg * CMID;
  const float* wv;
  float acc;
  if (o < 36) { wv = w_loc + (size_t)o * CMID;          acc = b_loc[o]; }
  else        { wv = w_score + (size_t)(o - 36) * CMID; acc = b_score[o - 36]; }
#pragma unroll 4
  for (int k = 0; k < CMID; ++k) acc += (float)fb[k] * wv[k];

  if (o < 36) out_loc[(size_t)n * (NBOX * 4) + (size_t)p * 36 + o] = acc;
  else        out_score[(size_t)n * (NBOX * 2) + (size_t)p * 18 + (o - 36)] = acc;
}

// fg = softmax(scores)[...,1] = sigmoid(s1 - s0)
__global__ void fg_kernel(const float* __restrict__ scores,
                          float* __restrict__ fg, int total) {
  int i = blockIdx.x * blockDim.x + threadIdx.x;
  if (i >= total) return;
  float s0 = scores[(size_t)i * 2 + 0];
  float s1 = scores[(size_t)i * 2 + 1];
  fg[i] = 1.f / (1.f + __expf(s0 - s1));
}

// Anchors (replicates reference base-anchor formula, incl. its h/w placement)
__global__ void anchors_kernel(float* __restrict__ anc, int total) {
  int i = blockIdx.x * blockDim.x + threadIdx.x;
  if (i >= total) return;
  int a = i % 9;
  int p = i / 9;
  int ri = a / 3, sj = a % 3;
  float r = (ri == 0) ? 0.5f : (ri == 1) ? 1.0f : 2.0f;
  float s = (sj == 0) ? 8.0f : (sj == 1) ? 16.0f : 32.0f;
  float h = 16.f * s * __fsqrt_rn(r);
  float w = 16.f * s * __fsqrt_rn(1.f / r);
  float sx = (float)((p & 63) << 4);
  float sy = (float)((p >> 6) << 4);
  anc[(size_t)i * 4 + 0] = sx - 0.5f * h;
  anc[(size_t)i * 4 + 1] = sy - 0.5f * w;
  anc[(size_t)i * 4 + 2] = sx + 0.5f * h;
  anc[(size_t)i * 4 + 3] = sy + 0.5f * w;
}

// ---------------------------------------------------------------------------
// Proposals + sequential NMS: one 1024-thread block per image.
// ---------------------------------------------------------------------------
__global__ __launch_bounds__(1024)
void proposals_kernel(const float* __restrict__ locs,
                      const float* __restrict__ fg,
                      const float* __restrict__ anc,
                      float* __restrict__ roi_buf,
                      float* __restrict__ s_buf,
                      int*   __restrict__ pick_buf,
                      float* __restrict__ out_rois,
                      float* __restrict__ out_inds,
                      const int* __restrict__ imgh_p,
                      const int* __restrict__ imgw_p) {
  int n   = blockIdx.x;
  int tid = threadIdx.x;
  float img_h = (float)imgh_p[0];
  float img_w = (float)imgw_p[0];

  const float* loc = locs + (size_t)n * NBOX * 4;
  const float* sc  = fg + (size_t)n * NBOX;
  float* roi  = roi_buf + (size_t)n * NBOX * 4;
  float* s    = s_buf + (size_t)n * NBOX;
  int*   pick = pick_buf + n * NPOST;

  __shared__ float rv[1024];
  __shared__ int   ridx[1024];
  __shared__ float bb[4];
  __shared__ int   bcount;
  if (tid == 0) bcount = 0;

  for (int i = tid; i < NBOX; i += 1024) {
    float a0 = anc[(size_t)i * 4 + 0], a1 = anc[(size_t)i * 4 + 1];
    float a2 = anc[(size_t)i * 4 + 2], a3 = anc[(size_t)i * 4 + 3];
    float aw = a2 - a0, ah = a3 - a1;
    float acx = a0 + 0.5f * aw, acy = a1 + 0.5f * ah;
    float l0 = loc[(size_t)i * 4 + 0], l1 = loc[(size_t)i * 4 + 1];
    float l2 = loc[(size_t)i * 4 + 2], l3 = loc[(size_t)i * 4 + 3];
    float cx = l0 * aw + acx, cy = l1 * ah + acy;
    float w = __expf(l2) * aw, h = __expf(l3) * ah;
    float x1 = fminf(fmaxf(cx - 0.5f * w, 0.f), img_w);
    float y1 = fminf(fmaxf(cy - 0.5f * h, 0.f), img_h);
    float x2 = fminf(fmaxf(cx + 0.5f * w, 0.f), img_w);
    float y2 = fminf(fmaxf(cy + 0.5f * h, 0.f), img_h);
    roi[(size_t)i * 4 + 0] = x1; roi[(size_t)i * 4 + 1] = y1;
    roi[(size_t)i * 4 + 2] = x2; roi[(size_t)i * 4 + 3] = y2;
    bool valid = (x2 - x1 >= 16.f) && (y2 - y1 >= 16.f);
    s[i] = valid ? sc[i] : -__builtin_inff();
  }
  __syncthreads();

  for (int it = 0; it < NPOST; ++it) {
    float best = -__builtin_inff();
    int bi = 0;
    for (int i = tid; i < NBOX; i += 1024) {
      float v = s[i];
      if (v > best) { best = v; bi = i; }
    }
    rv[tid] = best; ridx[tid] = bi;
    __syncthreads();
    for (int off = 512; off > 0; off >>= 1) {
      if (tid < off) {
        if (rv[tid + off] > rv[tid] ||
            (rv[tid + off] == rv[tid] && ridx[tid + off] < ridx[tid])) {
          rv[tid] = rv[tid + off]; ridx[tid] = ridx[tid + off];
        }
      }
      __syncthreads();
    }
    int  besti = ridx[0];
    bool ok = rv[0] > -__builtin_inff();
    if (tid == 0) {
      pick[it] = besti;
      if (ok) bcount = it + 1;  // ok picks form a prefix
      bb[0] = roi[(size_t)besti * 4 + 0]; bb[1] = roi[(size_t)besti * 4 + 1];
      bb[2] = roi[(size_t)besti * 4 + 2]; bb[3] = roi[(size_t)besti * 4 + 3];
      s[besti] = -__builtin_inff();
    }
    __syncthreads();
    if (ok) {
      float bx1 = bb[0], by1 = bb[1], bx2 = bb[2], by2 = bb[3];
      float barea = (bx2 - bx1) * (by2 - by1);
      for (int i = tid; i < NBOX; i += 1024) {
        float x1 = roi[(size_t)i * 4 + 0], y1 = roi[(size_t)i * 4 + 1];
        float x2 = roi[(size_t)i * 4 + 2], y2 = roi[(size_t)i * 4 + 3];
        float iw = fminf(bx2, x2) - fmaxf(bx1, x1); iw = iw > 0.f ? iw : 0.f;
        float ih = fminf(by2, y2) - fmaxf(by1, y1); ih = ih > 0.f ? ih : 0.f;
        float inter = iw * ih;
        float area = (x2 - x1) * (y2 - y1);
        float iou = inter / (barea + area - inter);
        if (iou > 0.7f) s[i] = -__builtin_inff();
      }
    }
    __syncthreads();
  }

  int cnt = bcount > 0 ? bcount : 1;
  for (int t = tid; t < NPOST; t += 1024) {
    int src = pick[t % cnt];
    out_rois[((size_t)n * NPOST + t) * 4 + 0] = roi[(size_t)src * 4 + 0];
    out_rois[((size_t)n * NPOST + t) * 4 + 1] = roi[(size_t)src * 4 + 1];
    out_rois[((size_t)n * NPOST + t) * 4 + 2] = roi[(size_t)src * 4 + 2];
    out_rois[((size_t)n * NPOST + t) * 4 + 3] = roi[(size_t)src * 4 + 3];
    out_inds[(size_t)n * NPOST + t] = (float)n;
  }
}

// ---------------------------------------------------------------------------
// Launch
// ---------------------------------------------------------------------------
extern "C" void kernel_launch(void* const* d_in, const int* in_sizes, int n_in,
                              void* d_out, int out_size, void* d_ws, size_t ws_size,
                              hipStream_t stream) {
  const float* x       = (const float*)d_in[0];
  const float* w1      = (const float*)d_in[1];
  const float* b1      = (const float*)d_in[2];
  const float* w_score = (const float*)d_in[3];
  const float* b_score = (const float*)d_in[4];
  const float* w_loc   = (const float*)d_in[5];
  const float* b_loc   = (const float*)d_in[6];
  const int*   imgh    = (const int*)d_in[7];
  const int*   imgw    = (const int*)d_in[8];

  float* out = (float*)d_out;
  float* out_locs   = out;                 // 8*36864*4 = 1,179,648
  float* out_scores = out + 1179648;       // 8*36864*2 =   589,824
  float* out_rois   = out + 1769472;       // 8*600*4   =    19,200
  float* out_inds   = out + 1788672;       // 8*600     =     4,800
  float* out_anc    = out + 1793472;       // 36864*4   =   147,456

  char* ws = (char*)d_ws;
  __bf16* xb   = (__bf16*)(ws + 0);                  // 35,684,352 B
  __bf16* wt   = (__bf16*)(ws + 35684352);           //  4,718,592 B
  __bf16* f    = (__bf16*)(ws + 40402944);           // 33,554,432 B
  float*  fg   = (float*)(ws + 73957376);            //  1,179,648 B
  float*  roib = (float*)(ws + 75137024);            //  4,718,592 B
  float*  sbuf = (float*)(ws + 79855616);            //  1,179,648 B
  int*    pick = (int*)(ws + 81035264);              //     19,200 B

  {
    int total = NB * HP * WP * CIN;                  // 17,842,176
    prep_x_kernel<<<(total + 255) / 256, 256, 0, stream>>>(x, xb, total);
  }
  {
    int total = 9 * CMID * CIN;                      // 2,359,296
    prep_w_kernel<<<(total + 255) / 256, 256, 0, stream>>>(w1, wt, total);
  }

  // WMMA conv + ReLU: 8 img * 64 rows * 4 col-tiles * 2 ch-groups = 4096
  conv3x3_wmma_kernel<<<NB * HH * 4 * 2, 128, 0, stream>>>(xb, wt, b1, f);

  {
    int total = NB * HH * WW * 64;                   // 2,097,152
    head1x1_kernel<<<(total + 255) / 256, 256, 0, stream>>>(
        f, w_loc, b_loc, w_score, b_score, out_locs, out_scores);
  }
  {
    int total = NB * NBOX;                           // 294,912
    fg_kernel<<<(total + 255) / 256, 256, 0, stream>>>(out_scores, fg, total);
  }
  {
    int total = NBOX;
    anchors_kernel<<<(total + 255) / 256, 256, 0, stream>>>(out_anc, total);
  }

  proposals_kernel<<<NB, 1024, 0, stream>>>(out_locs, fg, out_anc,
                                            roib, sbuf, pick,
                                            out_rois, out_inds, imgh, imgw);
}
